// LSTMPredictor_10625749090631
// MI455X (gfx1250) — matrix operand order repaired and checked by reading, not compiled, not run on hardware
//
#include <hip/hip_runtime.h>
#include <hip/hip_bf16.h>

#define BB 512
#define TT 1024
#define DD 64
#define HH 64
#define RR 4
#define MHH 32

typedef __attribute__((ext_vector_type(16))) __bf16 v16bf;
typedef __attribute__((ext_vector_type(2)))  __bf16 v2bf;
typedef __attribute__((ext_vector_type(8)))  float  v8f;

// ---- helpers -------------------------------------------------------------

// single fp32 -> bf16 (hardware cvt), as raw u16
static __device__ __forceinline__ unsigned short f2bfu(float f) {
  union { __bf16 b; unsigned short s; } u;
  u.b = (__bf16)f;
  return u.s;
}

// two fp32 -> packed bf16x2 (v_cvt_pk_bf16_f32 when available)
#if __has_builtin(__builtin_amdgcn_cvt_pk_bf16_f32)
static __device__ __forceinline__ unsigned cvt2(float lo, float hi) {
  union { v2bf v; unsigned u; } r;
  r.v = __builtin_amdgcn_cvt_pk_bf16_f32(lo, hi);
  return r.u;
}
#else
static __device__ __forceinline__ unsigned cvt2(float lo, float hi) {
  union { __bf16 b[2]; unsigned u; } r;
  r.b[0] = (__bf16)lo;
  r.b[1] = (__bf16)hi;
  return r.u;
}
#endif

// native v_tanh_f32 when available (gfx1250 TRANS op), else exp-based
#if __has_builtin(__builtin_amdgcn_tanhf)
static __device__ __forceinline__ float tanh_f(float x) {
  return __builtin_amdgcn_tanhf(x);
}
#else
static __device__ __forceinline__ float tanh_f(float x) {
  return 1.0f - 2.0f / (__expf(2.0f * x) + 1.0f);
}
#endif
static __device__ __forceinline__ float sigm(float x) {
  return fmaf(0.5f, tanh_f(0.5f * x), 0.5f);  // avoids the fp32 divide
}

// A-fragment (16x32 bf16): lane = M (0..15), kh = lane>>4.
// elements 0..7  : K = kbase + kh*16 + i
// elements 8..15 : K = kbase + kh*16 + i
static __device__ __forceinline__ v16bf fragA_lds(const unsigned short* p) {
  union { v16bf v; uint4 q[2]; } u;
  u.q[0] = ((const uint4*)p)[0];
  u.q[1] = ((const uint4*)(p + 8))[0];
  return u.v;
}
static __device__ __forceinline__ v16bf fragA_gf32(const float* p) {
  union { v16bf v; unsigned u[8]; } u;
#pragma unroll
  for (int i = 0; i < 4; ++i) u.u[i] = cvt2(p[2 * i], p[2 * i + 1]);
#pragma unroll
  for (int i = 0; i < 4; ++i) u.u[4 + i] = cvt2(p[8 + 2 * i], p[8 + 2 * i + 1]);
  return u.v;
}

// B-fragment (32x16 bf16): lane = N, kh = lane>>4; elements i: K = kbase + kh*16 + i
static __device__ __forceinline__ v16bf fragB_lds(const unsigned short* p) {
  union { v16bf v; uint4 q[2]; } u;
  u.q[0] = ((const uint4*)p)[0];
  u.q[1] = ((const uint4*)p)[1];
  return u.v;
}
static __device__ __forceinline__ v16bf fragB_gf32(const float* p) {
  union { v16bf v; unsigned u[8]; } u;
#pragma unroll
  for (int i = 0; i < 8; ++i) u.u[i] = cvt2(p[2 * i], p[2 * i + 1]);
  return u.v;
}

static __device__ __forceinline__ v8f wmma_bf16(v16bf a, v16bf b, v8f c) {
  return __builtin_amdgcn_wmma_f32_16x16x32_bf16(false, a, false, b, (short)0, c,
                                                 false, false);
}

// ---- kernel 1: recurrent LSTM, one block per 16-row batch tile -----------
// 4 waves; wave w owns hidden columns [16w, 16w+16) of all four gates, so the
// i/f/g/o combine and cell state c stay lane-local. x_t is staged to LDS once
// per step (shared by all 4 waves); h is exchanged via LDS; y rows are staged
// in LDS and stored with b128 vector stores.

__global__ __launch_bounds__(128) void lstm_kernel(
    const float* __restrict__ x, const float* __restrict__ h_init,
    const float* __restrict__ c_init, const float* __restrict__ W_ih,
    const float* __restrict__ W_hh, const float* __restrict__ b_ih,
    const float* __restrict__ b_hh, float* __restrict__ y) {
  __shared__ __align__(16) unsigned short hb[16 * HH];  // h tile, bf16 [m][j]
  __shared__ __align__(16) unsigned short xb[16 * DD];  // x_t tile, bf16 [m][d]
  __shared__ __align__(16) float yb[16 * HH];           // h tile, f32 (for y)

  const int tid  = threadIdx.x;
  const int lane = tid & 31;
  const int wave = tid >> 5;
  const int ln   = lane & 15;
  const int kh   = lane >> 4;
  const int bb   = blockIdx.x;  // batch tile index

  // ---- one-time: weight B-fragments into registers, biases -------------
  v16bf wih[4][2], whh[4][2];
  float bias[4];
#pragma unroll
  for (int g = 0; g < 4; ++g) {
    const int n = g * HH + wave * 16 + ln;  // gate row this lane owns
    bias[g] = b_ih[n] + b_hh[n];
#pragma unroll
    for (int kt = 0; kt < 2; ++kt) {
      const int kb = kt * 32 + kh * 16;
      wih[g][kt] = fragB_gf32(W_ih + (size_t)n * DD + kb);
      whh[g][kt] = fragB_gf32(W_hh + (size_t)n * HH + kb);
    }
  }

  const int j = wave * 16 + ln;  // hidden column owned in C/D fragments
  v8f c;
  {
    const float c0 = c_init[j];
#pragma unroll
    for (int v = 0; v < 8; ++v) c[v] = c0;
  }
  for (int idx = tid; idx < 16 * HH; idx += 128)
    hb[idx] = f2bfu(h_init[idx & (HH - 1)]);

  // staging role: thread covers row srow, cols [scol, scol+8)
  const int srow = tid >> 3;
  const int scol = (tid & 7) * 8;
  const float* xsrc = x + (size_t)(bb * 16 + srow) * TT * DD + scol;
  float* ydst = y + (size_t)(bb * 16 + srow) * TT * HH + scol;

  for (int t = 0; t < TT; ++t) {
    // ---- stage x_t into LDS as bf16 (2x b128 load, 4 packed cvt, b128 ds)
    {
      const float4 f0 = ((const float4*)(xsrc + (size_t)t * DD))[0];
      const float4 f1 = ((const float4*)(xsrc + (size_t)t * DD))[1];
      if (t + 1 < TT) __builtin_prefetch(xsrc + (size_t)(t + 1) * DD, 0, 1);
      uint4 pk;
      pk.x = cvt2(f0.x, f0.y);
      pk.y = cvt2(f0.z, f0.w);
      pk.z = cvt2(f1.x, f1.y);
      pk.w = cvt2(f1.z, f1.w);
      *(uint4*)(xb + srow * DD + scol) = pk;
    }
    __syncthreads();  // B1: xb ready; hb(t-1) ready

    // ---- A fragments from LDS
    v16bf ax[2], ah[2];
#pragma unroll
    for (int kt = 0; kt < 2; ++kt) {
      const int kb = kt * 32 + kh * 16;
      ax[kt] = fragA_lds(xb + ln * DD + kb);
      ah[kt] = fragA_lds(hb + ln * HH + kb);
    }

    v8f acc[4];
#pragma unroll
    for (int g = 0; g < 4; ++g) {
      v8f a;
#pragma unroll
      for (int v = 0; v < 8; ++v) a[v] = bias[g];
      a = wmma_bf16(ax[0], wih[g][0], a);
      a = wmma_bf16(ax[1], wih[g][1], a);
      a = wmma_bf16(ah[0], whh[g][0], a);
      a = wmma_bf16(ah[1], whh[g][1], a);
      acc[g] = a;
    }

    float hv[8];
#pragma unroll
    for (int v = 0; v < 8; ++v) {
      const float ig = sigm(acc[0][v]);
      const float fg = sigm(acc[1][v]);
      const float gg = tanh_f(acc[2][v]);
      const float og = sigm(acc[3][v]);
      const float cv = fg * c[v] + ig * gg;
      c[v] = cv;
      hv[v] = og * tanh_f(cv);
    }

    __syncthreads();  // B2: all waves finished reading hb/xb
#pragma unroll
    for (int v = 0; v < 8; ++v) {
      const int m = v + 8 * kh;  // C/D row
      hb[m * HH + j] = f2bfu(hv[v]);
      yb[m * HH + j] = hv[v];
    }
    __syncthreads();  // B3: hb/yb for step t complete

    // ---- cooperative y store: contiguous b128 stores
    {
      float* yo = ydst + (size_t)t * HH;
      ((float4*)yo)[0] = ((const float4*)(yb + srow * HH + scol))[0];
      ((float4*)yo)[1] = ((const float4*)(yb + srow * HH + scol))[1];
    }
  }
}

// ---- kernel 2: 4 stacked MLP heads over every (b,t) ----------------------
// 256 threads = 8 waves; each wave handles a 16-position tile.

__global__ __launch_bounds__(256) void heads_kernel(
    const float* __restrict__ y, const float* __restrict__ W1,
    const float* __restrict__ b1, const float* __restrict__ W2,
    const float* __restrict__ b2, const float* __restrict__ W3,
    const float* __restrict__ b3, float* __restrict__ out) {
  __shared__ __align__(16) unsigned short w1l[RR * MHH * HH];    // (128,64) bf16
  __shared__ __align__(16) unsigned short w2l[RR * MHH * MHH];   // 4x(32,32) bf16
  __shared__ __align__(16) unsigned short h1l[8][16 * RR * MHH]; // per-wave 16x128
  __shared__ float b1l[RR * MHH], b2l[RR * MHH], w3l[RR * MHH], b3l[RR];

  const int tid  = threadIdx.x;
  const int lane = tid & 31;
  const int wave = tid >> 5;
  const int ln   = lane & 15;
  const int kh   = lane >> 4;

  for (int i = tid; i < RR * MHH * HH; i += 256) w1l[i] = f2bfu(W1[i]);
  for (int i = tid; i < RR * MHH * MHH; i += 256) w2l[i] = f2bfu(W2[i]);
  for (int i = tid; i < RR * MHH; i += 256) {
    b1l[i] = b1[i];
    b2l[i] = b2[i];
    w3l[i] = W3[i];
  }
  if (tid < RR) b3l[tid] = b3[tid];
  __syncthreads();

  const size_t p0 = (size_t)blockIdx.x * 128 + (size_t)wave * 16;

  // layer 1: y(16x64) x W1^T(64x128), A from global fp32 -> bf16
  v16bf ay[2];
#pragma unroll
  for (int kt = 0; kt < 2; ++kt)
    ay[kt] = fragA_gf32(y + (p0 + ln) * HH + kt * 32 + kh * 16);

  unsigned short* h1 = h1l[wave];
#pragma unroll
  for (int nt = 0; nt < 8; ++nt) {
    const int n = nt * 16 + ln;
    v8f a;
    const float bv = b1l[n];
#pragma unroll
    for (int v = 0; v < 8; ++v) a[v] = bv;
    a = wmma_bf16(ay[0], fragB_lds(w1l + n * HH + kh * 16), a);
    a = wmma_bf16(ay[1], fragB_lds(w1l + n * HH + 32 + kh * 16), a);
#pragma unroll
    for (int v = 0; v < 8; ++v) {
      const int m = v + 8 * kh;
      h1[m * (RR * MHH) + n] = f2bfu(fmaxf(a[v], 0.0f));
    }
  }
  __syncthreads();

  // layers 2+3 per head: (16x32)x(32x32) WMMA, then w3-weighted shuffle reduce
#pragma unroll
  for (int r = 0; r < RR; ++r) {
    const v16bf a2 = fragA_lds(h1 + ln * (RR * MHH) + r * MHH + kh * 16);
    float res[8];
#pragma unroll
    for (int v = 0; v < 8; ++v) res[v] = 0.0f;
#pragma unroll
    for (int nt = 0; nt < 2; ++nt) {
      const int n = nt * 16 + ln;
      v8f a;
      const float bv = b2l[r * MHH + n];
#pragma unroll
      for (int v = 0; v < 8; ++v) a[v] = bv;
      a = wmma_bf16(a2, fragB_lds(w2l + (r * MHH + n) * MHH + kh * 16), a);
      const float wv = w3l[r * MHH + n];
#pragma unroll
      for (int v = 0; v < 8; ++v) {
        float t2 = fmaxf(a[v], 0.0f) * wv;   // h2[m][n] * w3[r][n]
        t2 += __shfl_xor(t2, 1, 32);         // reduce over the 16-lane N group
        t2 += __shfl_xor(t2, 2, 32);
        t2 += __shfl_xor(t2, 4, 32);
        t2 += __shfl_xor(t2, 8, 32);
        res[v] += t2;
      }
    }
    if (ln == 0) {
#pragma unroll
      for (int v = 0; v < 8; ++v) {
        const int m = v + 8 * kh;
        out[(size_t)r * BB * TT + p0 + m] = res[v] + b3l[r];
      }
    }
  }
}

// ---- launcher ------------------------------------------------------------

extern "C" void kernel_launch(void* const* d_in, const int* in_sizes, int n_in,
                              void* d_out, int out_size, void* d_ws,
                              size_t ws_size, hipStream_t stream) {
  (void)in_sizes; (void)n_in; (void)out_size; (void)d_ws; (void)ws_size;
  const float* x      = (const float*)d_in[0];
  const float* h_init = (const float*)d_in[1];
  const float* c_init = (const float*)d_in[2];
  const float* W_ih   = (const float*)d_in[3];
  const float* W_hh   = (const float*)d_in[4];
  const float* b_ih   = (const float*)d_in[5];
  const float* b_hh   = (const float*)d_in[6];
  const float* W1     = (const float*)d_in[7];
  const float* b1     = (const float*)d_in[8];
  const float* W2     = (const float*)d_in[9];
  const float* b2     = (const float*)d_in[10];
  const float* W3     = (const float*)d_in[11];
  const float* b3     = (const float*)d_in[12];

  float* y    = (float*)d_out;                       // (B,T,H)
  float* outs = y + (size_t)BB * TT * HH;            // (R,B,T)

  lstm_kernel<<<dim3(BB / 16), dim3(128), 0, stream>>>(
      x, h_init, c_init, W_ih, W_hh, b_ih, b_hh, y);
  heads_kernel<<<dim3((BB * TT) / 128), dim3(256), 0, stream>>>(
      y, W1, b1, W2, b2, W3, b3, outs);
}
